// FlexAttentionMHA_87041807220789
// MI455X (gfx1250) — matrix operand order repaired and checked by reading, not compiled, not run on hardware
//
#include <hip/hip_runtime.h>

// ---------------------------------------------------------------------------
// FlexAttention MHA for MI455X (gfx1250, wave32, WMMA).
// Pipeline: 3x GEMM (proj, fp32->bf16) -> 128x flash-attention chunks ->
//           1x GEMM (out proj, bf16->fp32). All matmuls on
//           v_wmma_f32_16x16x32_bf16. bf16 tile staging uses
//           GLOBAL_LOAD_ASYNC_TO_LDS_B128 (ASYNCcnt) when the builtin exists.
// ---------------------------------------------------------------------------

typedef __attribute__((ext_vector_type(16))) __bf16 bf16x16;
typedef __attribute__((ext_vector_type(4)))  __bf16 bf16x4;
typedef __attribute__((ext_vector_type(8)))  float  f32x8;
typedef __attribute__((ext_vector_type(4)))  int    i32x4;

#define DMODEL 1024
#define TOTALQ 8192
#define NHEADS 16
#define NSEQ   8
#define LSEQ   1024
#define DH     64
#define NCHUNK 128  /* NHEADS * NSEQ */

// ------------------------- async LDS staging -------------------------------
#if defined(__HIP_DEVICE_COMPILE__) && \
    __has_builtin(__builtin_amdgcn_global_load_async_to_lds_b128)
#define HAVE_ASYNC_LDS 1
#endif

// Copy 16 bytes (8 bf16) global -> LDS. Async (ASYNCcnt) when available,
// else through VGPRs. Builtin prototype (from clang diagnostic):
//   (int4 AS1*, int4 AS3*, imm i32 offset, imm i32 cpol)
__device__ inline void stage16(__bf16* dst_lds, const __bf16* src_glob) {
#ifdef HAVE_ASYNC_LDS
  __builtin_amdgcn_global_load_async_to_lds_b128(
      (__attribute__((address_space(1))) i32x4*)src_glob,
      (__attribute__((address_space(3))) i32x4*)dst_lds, 0, 0);
#else
  *(uint4*)dst_lds = *(const uint4*)src_glob;
#endif
}

__device__ inline void stage_fence() {
#ifdef HAVE_ASYNC_LDS
#if __has_builtin(__builtin_amdgcn_s_wait_asynccnt)
  __builtin_amdgcn_s_wait_asynccnt(0);
#else
  asm volatile("s_wait_asynccnt 0x0" ::: "memory");
#endif
#endif
}

// --------------------------- fragment loaders ------------------------------
// ISA 7.12.2, 16-bit A 16x32: lane<16 holds K[kb+0..7] then K[kb+16..23];
// lane>=16 holds K[kb+8..15] then K[kb+24..31] (row M = lane&15).
__device__ inline bf16x16 load_a_frag(const __bf16* smem, int stride,
                                      int m_base, int k_base, int lane) {
  const int m  = m_base + (lane & 15);
  const int hl = lane >> 4;
  const __bf16* p = smem + m * stride + k_base + hl * 8;
  union { bf16x16 v; uint4 q[2]; } u;
  u.q[0] = *(const uint4*)(p);
  u.q[1] = *(const uint4*)(p + 16);
  return u.v;
}

// 16-bit B 32x16 (stored as [n][k] in LDS): lane<16 holds K[kb+0..7] then K[kb+16..23],
// lane>=16 holds K[kb+8..15] then K[kb+24..31]; column N = lane&15.
__device__ inline bf16x16 load_b_frag(const __bf16* smem, int stride,
                                      int n_base, int k_base, int lane) {
  const int n  = n_base + (lane & 15);
  const int hl = lane >> 4;
  const __bf16* p = smem + n * stride + k_base + hl * 8;
  union { bf16x16 v; uint4 q[2]; } u;
  u.q[0] = *(const uint4*)(p);
  u.q[1] = *(const uint4*)(p + 16);
  return u.v;
}

__device__ inline f32x8 wmma_bf16(bf16x16 a, bf16x16 b, f32x8 c) {
  return __builtin_amdgcn_wmma_f32_16x16x32_bf16(
      /*neg_a=*/false, a, /*neg_b=*/false, b,
      /*c_mod=*/(short)0, c, /*reuse_a=*/false, /*reuse_b=*/false);
}

// --------------------------------- GEMM ------------------------------------
// C[M,N] = A[M,K] @ W[N,K]^T + bias.  M=8192, N=K=1024.
// Block: 128x128 tile, BK=32, 256 threads = 8 waves in a 2x4 grid,
// each wave owns a 64x32 patch = 4x2 WMMA accumulators.
template <bool A_BF16, bool OUT_BF16>
__global__ __launch_bounds__(256) void gemm_bt_kernel(
    const void* __restrict__ Av, const float* __restrict__ W,
    const float* __restrict__ bias, void* __restrict__ Cv) {
  constexpr int K = DMODEL, N = DMODEL, BM = 128, BN = 128, BK = 32, STR = 40;
  __shared__ __align__(16) __bf16 sA[BM * STR];
  __shared__ __align__(16) __bf16 sB[BN * STR];

  const int tid  = threadIdx.x;
  const int lane = tid & 31;
  const int wave = tid >> 5;
  const int bm = (blockIdx.x >> 3) * BM;
  const int bn = (blockIdx.x & 7) * BN;
  const int wm = (wave >> 2) * 64;  // 0 or 64
  const int wn = (wave & 3) * 32;   // 0,32,64,96

  f32x8 acc[4][2];
  for (int i = 0; i < 4; ++i)
    for (int j = 0; j < 2; ++j) acc[i][j] = (f32x8)(0.0f);

  for (int k0 = 0; k0 < K; k0 += BK) {
    // ---- stage A tile (async raw copy for bf16, VGPR convert for fp32) ----
    if constexpr (A_BF16) {
      const __bf16* A = (const __bf16*)Av;
      const int r = tid >> 2, c = (tid & 3) * 8;
      for (int p = 0; p < 2; ++p) {
        const int row = p * 64 + r;
        stage16(&sA[row * STR + c], A + (size_t)(bm + row) * K + k0 + c);
      }
    } else {
      const float* A = (const float*)Av;
      const int r = tid >> 3, c = (tid & 7) * 4;
      for (int p = 0; p < 4; ++p) {
        const int row = p * 32 + r;
        const float4 v = *(const float4*)(A + (size_t)(bm + row) * K + k0 + c);
        bf16x4 b;
        b.x = (__bf16)v.x; b.y = (__bf16)v.y;
        b.z = (__bf16)v.z; b.w = (__bf16)v.w;
        *(bf16x4*)&sA[row * STR + c] = b;
      }
    }
    // ---- stage W tile (always fp32 in global) ----
    {
      const int r = tid >> 3, c = (tid & 7) * 4;
      for (int p = 0; p < 4; ++p) {
        const int row = p * 32 + r;
        const float4 v = *(const float4*)(W + (size_t)(bn + row) * K + k0 + c);
        bf16x4 b;
        b.x = (__bf16)v.x; b.y = (__bf16)v.y;
        b.z = (__bf16)v.z; b.w = (__bf16)v.w;
        *(bf16x4*)&sB[row * STR + c] = b;
      }
    }
    if constexpr (A_BF16) stage_fence();
    __syncthreads();

    bf16x16 af[4], bfrg[2];
    for (int i = 0; i < 4; ++i) af[i] = load_a_frag(sA, STR, wm + i * 16, 0, lane);
    for (int j = 0; j < 2; ++j) bfrg[j] = load_b_frag(sB, STR, wn + j * 16, 0, lane);
    for (int i = 0; i < 4; ++i)
      for (int j = 0; j < 2; ++j) acc[i][j] = wmma_bf16(af[i], bfrg[j], acc[i][j]);
    __syncthreads();
  }

  // ---- epilogue: C layout, VGPR r -> row r + 8*half, lane&15 -> col ----
  const int ln = lane & 15, hl = lane >> 4;
  for (int i = 0; i < 4; ++i) {
    const int mrow = bm + wm + i * 16 + hl * 8;
    for (int j = 0; j < 2; ++j) {
      const int n = bn + wn + j * 16 + ln;
      const float bv = bias[n];
      for (int r = 0; r < 8; ++r) {
        const float val = acc[i][j][r] + bv;
        if constexpr (OUT_BF16)
          ((__bf16*)Cv)[(size_t)(mrow + r) * N + n] = (__bf16)val;
        else
          ((float*)Cv)[(size_t)(mrow + r) * N + n] = val;
      }
    }
  }
}

// ------------------------------ attention ----------------------------------
// One block = one (head,seq) chunk x 128 Q rows. 8 waves, wave w owns Q rows
// [w*16, w*16+16). Flash-attention over 16 KV tiles of 64 rows. Row<->lane
// mapping of WMMA C tiles is identical for S and O, so online-softmax stats
// live in registers; cross-row reduction = 16-lane shfl_xor.
__global__ __launch_bounds__(256) void attn_kernel(
    const __bf16* __restrict__ Qp, const __bf16* __restrict__ Kp,
    const __bf16* __restrict__ Vp, __bf16* __restrict__ Op) {
  constexpr int L = LSEQ, D = DH, BQ = 128, BKV = 64, STR = 72;
  const int chunk = blockIdx.x >> 3;
  const int qt    = blockIdx.x & 7;
  const int tid   = threadIdx.x;
  const int lane  = tid & 31;
  const int wave  = tid >> 5;
  const int ln    = lane & 15;
  const int hl    = lane >> 4;

  __shared__ __align__(16) __bf16 sQ[BQ * STR];
  __shared__ __align__(16) __bf16 sK[BKV * STR];
  __shared__ __align__(16) __bf16 sVt[D * STR];          // [dh][kv] transposed
  __shared__ __align__(16) __bf16 sP[8 * 16 * STR];      // per-wave P stripe

  const __bf16* Qc = Qp + (size_t)chunk * L * D + (size_t)qt * BQ * D;
  const __bf16* Kc = Kp + (size_t)chunk * L * D;
  const __bf16* Vc = Vp + (size_t)chunk * L * D;

  // ---- stage Q tile (128x64 bf16), async when available ----
  {
    const int r = tid >> 2, c = (tid & 3) * 16;
    for (int p = 0; p < 2; ++p) {
      const int row = p * 64 + r;
      stage16(&sQ[row * STR + c],     Qc + row * D + c);
      stage16(&sQ[row * STR + c + 8], Qc + row * D + c + 8);
    }
  }
  stage_fence();
  __syncthreads();

  bf16x16 aQ[2];
  aQ[0] = load_a_frag(sQ, STR, wave * 16, 0, lane);
  aQ[1] = load_a_frag(sQ, STR, wave * 16, 32, lane);

  float mprev[8], lprev[8];
  f32x8 o[4];
  for (int r = 0; r < 8; ++r) { mprev[r] = -1e30f; lprev[r] = 0.0f; }
  for (int t = 0; t < 4; ++t) o[t] = (f32x8)(0.0f);
  const float scale = 0.125f;  // 1/sqrt(64)

  for (int kt = 0; kt < L / BKV; ++kt) {
    const __bf16* Kt = Kc + (size_t)kt * BKV * D;
    const __bf16* Vt = Vc + (size_t)kt * BKV * D;
    // ---- stage K tile [64][64]: async copy (overlaps V transpose) ----
    {
      const int r = tid >> 2, c = (tid & 3) * 16;
      stage16(&sK[r * STR + c],     Kt + r * D + c);
      stage16(&sK[r * STR + c + 8], Kt + r * D + c + 8);
    }
    // ---- stage V tile transposed: sVt[dh][kv] (VGPR path, scatter) ----
    {
      const int r = tid >> 2, c = (tid & 3) * 16;
      for (int i = 0; i < 16; ++i) sVt[(c + i) * STR + r] = Vt[r * D + c + i];
    }
    stage_fence();
    __syncthreads();

    // ---- S = (Q K^T) * scale : 4 tiles of 16x16, reduce over dh=64 ----
    f32x8 s[4];
    for (int j = 0; j < 4; ++j) {
      f32x8 z = (f32x8)(0.0f);
      z = wmma_bf16(aQ[0], load_b_frag(sK, STR, j * 16, 0, lane), z);
      z = wmma_bf16(aQ[1], load_b_frag(sK, STR, j * 16, 32, lane), z);
      s[j] = z;
    }
    for (int j = 0; j < 4; ++j)
      for (int r = 0; r < 8; ++r) s[j][r] *= scale;

    // ---- online softmax ----
    float mnew[8], alpha[8];
    for (int r = 0; r < 8; ++r) {
      float v = fmaxf(fmaxf(s[0][r], s[1][r]), fmaxf(s[2][r], s[3][r]));
      for (int msk = 8; msk >= 1; msk >>= 1) v = fmaxf(v, __shfl_xor(v, msk));
      mnew[r]  = fmaxf(mprev[r], v);
      alpha[r] = __expf(mprev[r] - mnew[r]);
    }
    for (int j = 0; j < 4; ++j)
      for (int r = 0; r < 8; ++r) s[j][r] = __expf(s[j][r] - mnew[r]);
    for (int r = 0; r < 8; ++r) {
      float v = s[0][r] + s[1][r] + s[2][r] + s[3][r];
      for (int msk = 8; msk >= 1; msk >>= 1) v += __shfl_xor(v, msk);
      lprev[r] = lprev[r] * alpha[r] + v;
      mprev[r] = mnew[r];
    }
    for (int t = 0; t < 4; ++t)
      for (int r = 0; r < 8; ++r) o[t][r] *= alpha[r];

    // ---- P: C-layout -> A-layout via wave-private LDS stripe ----
    __bf16* pw = &sP[wave * 16 * STR];
    for (int j = 0; j < 4; ++j)
      for (int r = 0; r < 8; ++r)
        pw[(r + 8 * hl) * STR + j * 16 + ln] = (__bf16)s[j][r];

    // ---- O += P @ V : 4 dh-tiles, reduce over kv=64 ----
    for (int kc = 0; kc < 2; ++kc) {
      const bf16x16 aP = load_a_frag(pw, STR, 0, kc * 32, lane);
      for (int t = 0; t < 4; ++t)
        o[t] = wmma_bf16(aP, load_b_frag(sVt, STR, t * 16, kc * 32, lane), o[t]);
    }
    __syncthreads();
  }

  // ---- normalize & store (same flat layout as Qp) ----
  const size_t obase = (size_t)chunk * L * D + (size_t)qt * BQ * D;
  for (int r = 0; r < 8; ++r) {
    const float inv = 1.0f / lprev[r];
    const int row = wave * 16 + r + 8 * hl;
    for (int t = 0; t < 4; ++t)
      Op[obase + (size_t)row * D + t * 16 + ln] = (__bf16)(o[t][r] * inv);
  }
}

// ------------------------------ launcher -----------------------------------
extern "C" void kernel_launch(void* const* d_in, const int* in_sizes, int n_in,
                              void* d_out, int out_size, void* d_ws,
                              size_t ws_size, hipStream_t stream) {
  const float* query = (const float*)d_in[0];
  const float* key   = (const float*)d_in[1];
  const float* value = (const float*)d_in[2];
  const float* Wq = (const float*)d_in[5];
  const float* bq = (const float*)d_in[6];
  const float* Wk = (const float*)d_in[7];
  const float* bk = (const float*)d_in[8];
  const float* Wv = (const float*)d_in[9];
  const float* bv = (const float*)d_in[10];
  const float* Wo = (const float*)d_in[11];
  const float* bo = (const float*)d_in[12];
  float* out = (float*)d_out;

  const size_t elems = (size_t)TOTALQ * DMODEL;
  __bf16* Qp = (__bf16*)d_ws;
  __bf16* Kp = Qp + elems;
  __bf16* Vp = Kp + elems;
  __bf16* Ap = Vp + elems;

  dim3 blk(256);
  dim3 ggrid((TOTALQ / 128) * (DMODEL / 128));  // 512 blocks

  gemm_bt_kernel<false, true><<<ggrid, blk, 0, stream>>>(query, Wq, bq, Qp);
  gemm_bt_kernel<false, true><<<ggrid, blk, 0, stream>>>(key, Wk, bk, Kp);
  gemm_bt_kernel<false, true><<<ggrid, blk, 0, stream>>>(value, Wv, bv, Vp);
  attn_kernel<<<dim3(NCHUNK * (LSEQ / 128)), blk, 0, stream>>>(Qp, Kp, Vp, Ap);
  gemm_bt_kernel<true, false><<<ggrid, blk, 0, stream>>>(Ap, Wo, bo, out);
}
